// BiLevelRoutingAttention_nchw_51934744543817
// MI455X (gfx1250) — matrix operand order repaired and checked by reading, not compiled, not run on hardware
//
#include <hip/hip_runtime.h>
#include <hip/hip_bf16.h>

// ---------------------------------------------------------------------------
// BiLevelRoutingAttention (BiFormer) for MI455X / gfx1250.
// B=16, DIM=256, H=W=56, HEADS=8 (head_dim=32), NWIN=7 (49 regions of 8x8=64
// tokens), TOPK=4, SCALE = 1/16.
// All three GEMM stages use v_wmma_f32_16x16x32_f16 (wave32 WMMA).
// Fragment operands are fed with contiguous 128-bit LDS loads (ds_load_b128):
//  - A layout (16x32 MxK): lane reads rows [m][khi..khi+8) and [m][khi+16..+24)
//  - B layout (32x16 KxN): operand stored transposed [n][k]; lane reads
//    [n][kb..kb+8) and [n][kb+16..+24) as two b128s.
// ---------------------------------------------------------------------------

typedef __attribute__((ext_vector_type(16))) _Float16 v16h;
typedef __attribute__((ext_vector_type(8)))  _Float16 v8h;
typedef __attribute__((ext_vector_type(4)))  _Float16 v4h;
typedef __attribute__((ext_vector_type(8)))  float    v8f;

union V16H { v16h v; v8h h[2]; };

#define BATCH   16
#define DIM     256
#define HH      56
#define WW      56
#define HW      3136
#define HEADS   8
#define HDIM    32
#define NWIN    7
#define NREG    49
#define RTOK    64
#define TOPK    4
#define KTOT    256
#define SCALE   0.0625f

// row pitches (halves) chosen so rows are 16-byte aligned
#define PK32    40     // for [*][32]-K tiles: 40*2 = 80 B rows
#define PK256   264    // for [*][256]-K tiles: 264*2 = 528 B rows

// ======================= GEMM: Y[b] = W[M,256] @ X[b][256,3136] + bias ======
#define GM 128
#define GN 64
#define GK 32

__global__ __launch_bounds__(256) void gemm_nchw(
    const float* __restrict__ Wt,    // [M, 256]
    const float* __restrict__ bias,  // [M]
    const float* __restrict__ X,     // [B, 256, 3136]
    float* __restrict__ Y,           // [B, M, 3136]
    int M)
{
    const int bm = blockIdx.x;
    const int bn = blockIdx.y;
    const int b  = blockIdx.z;
    const float* __restrict__ Xb = X + (size_t)b * 256 * HW;
    float* __restrict__ Yb = Y + (size_t)b * M * HW;
    const int m0 = bm * GM;
    const int n0 = bn * GN;

    __shared__ __align__(16) _Float16 sW [GM][PK32];  // [m][k]  10 KB
    __shared__ __align__(16) _Float16 sXT[GN][PK32];  // [n][k]  5 KB (transposed)

    const int t    = threadIdx.x;
    const int wave = t >> 5;
    const int lane = t & 31;
    const int lan16 = lane & 15;
    const int grp  = lane >> 4;
    const int khi  = grp << 3;    // A: second k-octet selector
    const int kb   = grp << 3;    // B: second k-octet selector

    v8f acc[4];
#pragma unroll
    for (int i = 0; i < 4; ++i) acc[i] = (v8f){0,0,0,0,0,0,0,0};

    for (int k0 = 0; k0 < 256; k0 += GK) {
        // ---- W tile 128x32: 1024 float4 loads, packed v4h LDS stores ----
#pragma unroll
        for (int it = 0; it < 4; ++it) {
            const int lin = t + (it << 8);          // 0..1023
            const int r = lin >> 3;
            const int c4 = (lin & 7) << 2;
            float4 w4 = *(const float4*)&Wt[(size_t)(m0 + r) * 256 + (k0 + c4)];
            v4h h4 = { (_Float16)w4.x, (_Float16)w4.y,
                       (_Float16)w4.z, (_Float16)w4.w };
            *(v4h*)&sW[r][c4] = h4;
        }
        // ---- X tile 32x64 -> transposed [n][k] ----
#pragma unroll
        for (int it = 0; it < 2; ++it) {
            const int lin = t + (it << 8);          // 0..511
            const int r = lin >> 4;                 // k row
            const int c4 = (lin & 15) << 2;         // n col
            float4 x4 = *(const float4*)&Xb[(size_t)(k0 + r) * HW + (n0 + c4)];
            sXT[c4 + 0][r] = (_Float16)x4.x;
            sXT[c4 + 1][r] = (_Float16)x4.y;
            sXT[c4 + 2][r] = (_Float16)x4.z;
            sXT[c4 + 3][r] = (_Float16)x4.w;
        }
        __syncthreads();

        V16H a;
        {
            const int m = (wave << 4) + lan16;
            a.h[0] = *(const v8h*)&sW[m][khi];       // k = khi..khi+7
            a.h[1] = *(const v8h*)&sW[m][khi + 16];  // k = khi+16..khi+23
        }
#pragma unroll
        for (int nt = 0; nt < 4; ++nt) {
            const int n = (nt << 4) + lan16;
            V16H bfr;
            bfr.h[0] = *(const v8h*)&sXT[n][kb];
            bfr.h[1] = *(const v8h*)&sXT[n][kb + 16];
            acc[nt] = __builtin_amdgcn_wmma_f32_16x16x32_f16(
                false, a.v, false, bfr.v, (short)0, acc[nt], false, false);
        }
        __syncthreads();
    }

    const int mrow = m0 + (wave << 4) + (grp << 3);
    float bv[8];
#pragma unroll
    for (int j = 0; j < 8; ++j) bv[j] = bias[mrow + j];
#pragma unroll
    for (int nt = 0; nt < 4; ++nt) {
        const int n = n0 + (nt << 4) + lan16;
#pragma unroll
        for (int j = 0; j < 8; ++j)
            Yb[(size_t)(mrow + j) * HW + n] = acc[nt][j] + bv[j];
    }
}

// ======================= region pooling (q,k) ==============================
__global__ __launch_bounds__(64) void pool_kernel(
    const float* __restrict__ qkv, float* __restrict__ pool)
{
    const int bc = blockIdx.x;
    const int b = bc >> 9, c = bc & 511;
    const int r = threadIdx.x;
    if (r >= NREG) return;
    const int rh = r / NWIN, rw = r % NWIN;
    const float* __restrict__ src = qkv + ((size_t)b * 768 + c) * HW
                                  + (rh * 8) * WW + rw * 8;
    float s = 0.f;
#pragma unroll
    for (int pp = 0; pp < 8; ++pp) {
        float4 a0 = *(const float4*)(src + pp * WW);
        float4 a1 = *(const float4*)(src + pp * WW + 4);
        s += a0.x + a0.y + a0.z + a0.w + a1.x + a1.y + a1.z + a1.w;
    }
    pool[((size_t)b * 512 + c) * NREG + r] = s * (1.0f / 64.0f);
}

// ======================= routing: a_r = q_r @ k_r, top-4 ====================
__global__ __launch_bounds__(64) void router_kernel(
    const float* __restrict__ pool, int* __restrict__ idx)
{
    const int bq = blockIdx.x;
    const int b = bq / NREG, qr = bq % NREG;
    const int t = threadIdx.x;
    __shared__ float areg[NREG];
    if (t < NREG) {
        const float* __restrict__ qp = pool + (size_t)b * 512 * NREG;
        const float* __restrict__ kp = qp + 256 * NREG;
        float s = 0.f;
        for (int c = 0; c < 256; ++c)
            s += qp[c * NREG + qr] * kp[c * NREG + t];
        areg[t] = s;
    }
    __syncthreads();
    if (t == 0) {
        int sel[TOPK];
#pragma unroll
        for (int k = 0; k < TOPK; ++k) {
            float best = -3.4e38f; int bi = 0;
            for (int j = 0; j < NREG; ++j) {
                bool used = false;
                for (int p = 0; p < k; ++p) used |= (sel[p] == j);
                if (!used && areg[j] > best) { best = areg[j]; bi = j; }
            }
            sel[k] = bi;
            idx[((size_t)b * NREG + qr) * TOPK + k] = bi;
        }
    }
}

// ======================= fused attention per (b, head, region) ==============
struct SmemQK { _Float16 q[RTOK][PK32]; _Float16 k[KTOT][PK32]; };
union  SmemU  { SmemQK qk; _Float16 p[RTOK][PK256]; };

__global__ __launch_bounds__(256) void attn_kernel(
    const float* __restrict__ qkv,   // [B,768,3136]
    const int*   __restrict__ idx,   // [B,49,4]
    float*       __restrict__ out)   // [B,256,3136]
{
    const int region = blockIdx.x;
    const int head   = blockIdx.y;
    const int b      = blockIdx.z;
    const int rh = region / NWIN, rw = region % NWIN;
    const int base_sp = (rh * 8) * WW + rw * 8;

    __shared__ __align__(16) SmemU    su;                   // 33 KB
    __shared__ __align__(16) _Float16 sVT[HDIM][PK256];     // [d][token] 16.5 KB
    __shared__ float red[RTOK][2][2];
    __shared__ int   sidx[TOPK];

    const int t = threadIdx.x;
    const int wave = t >> 5;
    const int lane = t & 31;
    const int lan16 = lane & 15;
    const int grp = lane >> 4;
    const int khi = grp << 3;
    const int kb  = grp << 3;

    const size_t qoff = ((size_t)b * 768 + head * HDIM) * HW;
    const size_t koff = qoff + (size_t)256 * HW;
    const size_t voff = qoff + (size_t)512 * HW;

    if (t < TOPK) sidx[t] = idx[((size_t)b * NREG + region) * TOPK + t];
    __syncthreads();

    // ---- stage Q (scaled) into [p][d]: thread -> (d, region-row) ----
    {
        const int d = t >> 3, pr = t & 7;
        const float* src = qkv + qoff + (size_t)d * HW + base_sp + pr * WW;
        float4 a0 = *(const float4*)src;
        float4 a1 = *(const float4*)(src + 4);
        const int p = pr << 3;
        su.qk.q[p + 0][d] = (_Float16)(a0.x * SCALE);
        su.qk.q[p + 1][d] = (_Float16)(a0.y * SCALE);
        su.qk.q[p + 2][d] = (_Float16)(a0.z * SCALE);
        su.qk.q[p + 3][d] = (_Float16)(a0.w * SCALE);
        su.qk.q[p + 4][d] = (_Float16)(a1.x * SCALE);
        su.qk.q[p + 5][d] = (_Float16)(a1.y * SCALE);
        su.qk.q[p + 6][d] = (_Float16)(a1.z * SCALE);
        su.qk.q[p + 7][d] = (_Float16)(a1.w * SCALE);
    }
    // ---- gathered K into [token][d], V into transposed [d][token] ----
#pragma unroll
    for (int it = 0; it < 4; ++it) {
        const int lin = t + (it << 8);          // 0..1023
        const int kt = lin >> 8, rem = lin & 255;
        const int d = rem >> 3, pr = rem & 7;
        const int reg = sidx[kt];
        const int rsp = ((reg / NWIN) * 8 + pr) * WW + (reg % NWIN) * 8;
        {
            const float* src = qkv + koff + (size_t)d * HW + rsp;
            float4 a0 = *(const float4*)src;
            float4 a1 = *(const float4*)(src + 4);
            const int tok = kt * RTOK + (pr << 3);
            su.qk.k[tok + 0][d] = (_Float16)a0.x;
            su.qk.k[tok + 1][d] = (_Float16)a0.y;
            su.qk.k[tok + 2][d] = (_Float16)a0.z;
            su.qk.k[tok + 3][d] = (_Float16)a0.w;
            su.qk.k[tok + 4][d] = (_Float16)a1.x;
            su.qk.k[tok + 5][d] = (_Float16)a1.y;
            su.qk.k[tok + 6][d] = (_Float16)a1.z;
            su.qk.k[tok + 7][d] = (_Float16)a1.w;
        }
        {
            const float* src = qkv + voff + (size_t)d * HW + rsp;
            float4 a0 = *(const float4*)src;
            float4 a1 = *(const float4*)(src + 4);
            v8h h8 = { (_Float16)a0.x, (_Float16)a0.y, (_Float16)a0.z,
                       (_Float16)a0.w, (_Float16)a1.x, (_Float16)a1.y,
                       (_Float16)a1.z, (_Float16)a1.w };
            *(v8h*)&sVT[d][kt * RTOK + (pr << 3)] = h8;   // one ds_store_b128
        }
    }
    __syncthreads();

    // ---- S = Q K^T : wave w -> rows mi*16, columns nhalf*128 ----
    const int mi = wave >> 1;
    const int nhalf = wave & 1;
    const int m = (mi << 4) + lan16;
    v8f s[8];
    {
        V16H a;
        a.h[0] = *(const v8h*)&su.qk.q[m][khi];
        a.h[1] = *(const v8h*)&su.qk.q[m][khi + 16];
#pragma unroll
        for (int nt = 0; nt < 8; ++nt) {
            const int tok = ((nhalf << 3) + nt) * 16 + lan16;
            V16H bfr;
            bfr.h[0] = *(const v8h*)&su.qk.k[tok][kb];
            bfr.h[1] = *(const v8h*)&su.qk.k[tok][kb + 16];
            v8f z = (v8f){0,0,0,0,0,0,0,0};
            s[nt] = __builtin_amdgcn_wmma_f32_16x16x32_f16(
                false, a.v, false, bfr.v, (short)0, z, false, false);
        }
    }

    // ---- softmax over 256 columns (wave pairs share via LDS) ----
#pragma unroll
    for (int j = 0; j < 8; ++j) {
        float lm = s[0][j];
#pragma unroll
        for (int nt = 1; nt < 8; ++nt) lm = fmaxf(lm, s[nt][j]);
#pragma unroll
        for (int msk = 1; msk < 16; msk <<= 1)
            lm = fmaxf(lm, __shfl_xor(lm, msk, 32));
        if (lan16 == 0) red[(mi << 4) + j + (grp << 3)][nhalf][0] = lm;
    }
    __syncthreads();
#pragma unroll
    for (int j = 0; j < 8; ++j) {
        const int row = (mi << 4) + j + (grp << 3);
        const float gmax = fmaxf(red[row][0][0], red[row][1][0]);
        float ls = 0.f;
#pragma unroll
        for (int nt = 0; nt < 8; ++nt) {
            float e = __expf(s[nt][j] - gmax);
            s[nt][j] = e;
            ls += e;
        }
#pragma unroll
        for (int msk = 1; msk < 16; msk <<= 1)
            ls += __shfl_xor(ls, msk, 32);
        if (lan16 == 0) red[row][nhalf][1] = ls;
    }
    __syncthreads();
#pragma unroll
    for (int j = 0; j < 8; ++j) {
        const int row = (mi << 4) + j + (grp << 3);
        const float inv = 1.0f / (red[row][0][1] + red[row][1][1]);
#pragma unroll
        for (int nt = 0; nt < 8; ++nt) {
            const int col = ((nhalf << 3) + nt) * 16 + lan16;
            su.p[row][col] = (_Float16)(s[nt][j] * inv);
        }
    }
    __syncthreads();

    // ---- O = P V : wave -> tile (mi, nj), K-loop of 8 x 32 tokens ----
    const int nj = wave & 1;
    const int dcol = (nj << 4) + lan16;
    v8f o = (v8f){0,0,0,0,0,0,0,0};
#pragma unroll
    for (int kc = 0; kc < 8; ++kc) {
        V16H a, bfr;
        a.h[0]   = *(const v8h*)&su.p[m][(kc << 5) + khi];
        a.h[1]   = *(const v8h*)&su.p[m][(kc << 5) + khi + 16];
        bfr.h[0] = *(const v8h*)&sVT[dcol][(kc << 5) + kb];
        bfr.h[1] = *(const v8h*)&sVT[dcol][(kc << 5) + kb + 16];
        o = __builtin_amdgcn_wmma_f32_16x16x32_f16(
            false, a.v, false, bfr.v, (short)0, o, false, false);
    }

    // ---- seq2grid store ----
    float* __restrict__ op = out + ((size_t)b * DIM + head * HDIM + dcol) * HW;
#pragma unroll
    for (int j = 0; j < 8; ++j) {
        const int tok = (mi << 4) + j + (grp << 3);
        op[base_sp + (tok >> 3) * WW + (tok & 7)] = o[j];
    }
}

// ======================= LePE: depthwise 3x3 on v, accumulate ===============
__global__ __launch_bounds__(256) void lepe_kernel(
    const float* __restrict__ qkv, const float* __restrict__ lw,
    const float* __restrict__ lb, float* __restrict__ out)
{
    const int bc = blockIdx.x;
    const int b = bc >> 8, c = bc & 255;
    const float* __restrict__ v = qkv + ((size_t)b * 768 + 512 + c) * HW;
    float* __restrict__ o = out + ((size_t)b * DIM + c) * HW;
    float w9[9];
#pragma unroll
    for (int i = 0; i < 9; ++i) w9[i] = lw[c * 9 + i];
    const float bias = lb[c];

    for (int i = threadIdx.x; i < HW; i += 256) {
        const int h = i / WW, w = i % WW;
        float acc = bias;
#pragma unroll
        for (int kh = 0; kh < 3; ++kh) {
            const int hh = h + kh - 1;
            if (hh < 0 || hh >= HH) continue;
#pragma unroll
            for (int kw = 0; kw < 3; ++kw) {
                const int ww2 = w + kw - 1;
                if (ww2 < 0 || ww2 >= WW) continue;
                acc += v[hh * WW + ww2] * w9[kh * 3 + kw];
            }
        }
        o[i] += acc;
    }
}

// ======================= launch =============================================
extern "C" void kernel_launch(void* const* d_in, const int* in_sizes, int n_in,
                              void* d_out, int out_size, void* d_ws, size_t ws_size,
                              hipStream_t stream)
{
    const float* x      = (const float*)d_in[0];
    const float* qkv_w  = (const float*)d_in[1];
    const float* qkv_b  = (const float*)d_in[2];
    const float* lepe_w = (const float*)d_in[3];
    const float* lepe_b = (const float*)d_in[4];
    const float* out_w  = (const float*)d_in[5];
    const float* out_b  = (const float*)d_in[6];

    const size_t QKV_E  = (size_t)BATCH * 768 * HW;
    const size_t ATTN_E = (size_t)BATCH * DIM * HW;
    const size_t POOL_E = (size_t)BATCH * 512 * NREG;

    float* ws_qkv  = (float*)d_ws;
    float* ws_attn = ws_qkv + QKV_E;
    float* ws_pool = ws_attn + ATTN_E;
    int*   ws_idx  = (int*)(ws_pool + POOL_E);

    gemm_nchw<<<dim3(768 / GM, HW / GN, BATCH), 256, 0, stream>>>(
        qkv_w, qkv_b, x, ws_qkv, 768);

    pool_kernel<<<BATCH * 512, 64, 0, stream>>>(ws_qkv, ws_pool);

    router_kernel<<<BATCH * NREG, 64, 0, stream>>>(ws_pool, ws_idx);

    attn_kernel<<<dim3(NREG, HEADS, BATCH), 256, 0, stream>>>(
        ws_qkv, ws_idx, ws_attn);

    lepe_kernel<<<BATCH * DIM, 256, 0, stream>>>(
        ws_qkv, lepe_w, lepe_b, ws_attn);

    gemm_nchw<<<dim3(DIM / GM, HW / GN, BATCH), 256, 0, stream>>>(
        out_w, out_b, ws_attn, (float*)d_out, 256);
}
